// ExplainGNN_41987600286245
// MI455X (gfx1250) — matrix operand order repaired and checked by reading, not compiled, not run on hardware
//
#include <hip/hip_runtime.h>
#include <hip/hip_bf16.h>
#include <stdint.h>

// ExplainGNN fused pipeline for MI455X (gfx1250, wave32, WMMA).
// Dominant op: 12288x12288x256 pairwise-distance GEMM (77 GFLOP) done with
// v_wmma_f32_16x16x32_bf16 in split-precision (hi+lo bf16, 3 WMMA / K32 step),
// fused with segment-max reductions so the 604MB edge matrix is never stored.
// Staging uses GLOBAL_LOAD_ASYNC_TO_LDS_B128 + ASYNCcnt double-buffering when
// the toolchain exposes the builtins (guarded; sync fallback otherwise).
// Workspace required ~78 MB.

#define EN   12288
#define EL   12288
#define BN   512
#define BL   512
#define DIMD 256
#define KTOP 10
#define TILE 128
#define EMP  132          // padded LDS stride for the 128x128 em tile
#define ALPHA_C 0.5f
#define TEMP_C  0.5f

typedef __attribute__((ext_vector_type(16))) __bf16 v16bf;
typedef __attribute__((ext_vector_type(8)))  float  v8f;
typedef int v4i_vs __attribute__((vector_size(16)));   // matches builtin's V4i

#if defined(__AMDGCN__) && __has_builtin(__builtin_amdgcn_global_load_async_to_lds_b128) && __has_builtin(__builtin_amdgcn_s_wait_asynccnt)
#define USE_ASYNC_LDS 1
#else
#define USE_ASYNC_LDS 0
#endif

__device__ __forceinline__ unsigned f_as_u(float f) { union { float f; unsigned u; } c; c.f = f; return c.u; }
__device__ __forceinline__ float    u_as_f(unsigned u) { union { float f; unsigned u; } c; c.u = u; return c.f; }
#define NEG_INF_F u_as_f(0xFF800000u)

// Async global->LDS 16B copy (no VGPR round-trip, tracked by ASYNCcnt).
// Builtin prototype (from hipcc diagnostic): (v4i AS1*, v4i AS3*, imm int, imm int)
__device__ __forceinline__ void async_b128(const void* g, void* l)
{
#if USE_ASYNC_LDS
  __builtin_amdgcn_global_load_async_to_lds_b128(
      (__attribute__((address_space(1))) v4i_vs*)g,
      (__attribute__((address_space(3))) v4i_vs*)l, 0, 0);
#else
  *(uint4*)l = *(const uint4*)g;
#endif
}
template <int N> __device__ __forceinline__ void wait_async()
{
#if USE_ASYNC_LDS
  __builtin_amdgcn_s_wait_asynccnt((short)N);
#endif
}

// ---------------------------------------------------------------------------
// Kernel 1: edge features -> bf16 hi/lo split + squared norms.
// One wave32 per edge row; lane covers 8 of the 256 dims.
// ---------------------------------------------------------------------------
__global__ __launch_bounds__(256) void build_edge_feats(
    const float* __restrict__ emb,
    const int* __restrict__ node_edge, const int* __restrict__ label_edge,
    __bf16* __restrict__ efn_hi, __bf16* __restrict__ efn_lo,
    __bf16* __restrict__ efl_hi, __bf16* __restrict__ efl_lo,
    float* __restrict__ sqn, float* __restrict__ sql)
{
  int lane = threadIdx.x & 31;
  int row  = blockIdx.x * 8 + (threadIdx.x >> 5);
  const int* edge; __bf16* dh; __bf16* dl; float* dn; int e;
  if (row < EN) { edge = node_edge;  dh = efn_hi; dl = efn_lo; dn = sqn; e = row; }
  else          { edge = label_edge; dh = efl_hi; dl = efl_lo; dn = sql; e = row - EN; }
  int e0 = edge[e], e1 = edge[EN + e];   // [2, E] flat
  const float4* pa = (const float4*)(emb + (size_t)e0 * DIMD + lane * 8);
  const float4* pb = (const float4*)(emb + (size_t)e1 * DIMD + lane * 8);
  float4 a0 = pa[0], a1 = pa[1], b0 = pb[0], b1 = pb[1];
  float x[8] = {0.5f*(a0.x+b0.x), 0.5f*(a0.y+b0.y), 0.5f*(a0.z+b0.z), 0.5f*(a0.w+b0.w),
                0.5f*(a1.x+b1.x), 0.5f*(a1.y+b1.y), 0.5f*(a1.z+b1.z), 0.5f*(a1.w+b1.w)};
  union { __bf16 h[8]; uint4 u; } hi, lo;
  float sq = 0.f;
  #pragma unroll
  for (int i = 0; i < 8; i++) {
    sq += x[i] * x[i];
    __bf16 h = (__bf16)x[i];
    hi.h[i] = h;
    lo.h[i] = (__bf16)(x[i] - (float)h);
  }
  *(uint4*)(dh + (size_t)e * DIMD + lane * 8) = hi.u;
  *(uint4*)(dl + (size_t)e * DIMD + lane * 8) = lo.u;
  #pragma unroll
  for (int m = 16; m; m >>= 1) sq += __shfl_xor(sq, m, 32);
  if (lane == 0) dn[e] = sq;
}

// ---------------------------------------------------------------------------
// Kernel 2: segment bounds via binary search (seg arrays are sorted).
// ---------------------------------------------------------------------------
__global__ void seg_bounds(const int* __restrict__ node_seg, const int* __restrict__ label_seg,
                           int* startN, int* cntN, int* startL, int* cntL)
{
  const int* a; int* st; int* ct; int n;
  if (blockIdx.x == 0) { a = node_seg;  st = startN; ct = cntN; n = EN; }
  else                 { a = label_seg; st = startL; ct = cntL; n = EL; }
  int s = threadIdx.x;
  if (s < BN) {
    int lo = 0, hi = n;
    while (lo < hi) { int mid = (lo + hi) >> 1; if (a[mid] < s) lo = mid + 1; else hi = mid; }
    int lo2 = lo, hi2 = n;
    while (lo2 < hi2) { int mid = (lo2 + hi2) >> 1; if (a[mid] < s + 1) lo2 = mid + 1; else hi2 = mid; }
    st[s] = lo; ct[s] = lo2 - lo;
  }
}

// ---------------------------------------------------------------------------
// Kernel 3: fill rowmax+colmax with -inf bit pattern (identity for seg-max).
// ---------------------------------------------------------------------------
__global__ void init_neginf(uint4* __restrict__ p, int n4)
{
  uint4 v; v.x = v.y = v.z = v.w = 0xFF800000u;
  for (int i = blockIdx.x * blockDim.x + threadIdx.x; i < n4; i += gridDim.x * blockDim.x) p[i] = v;
}

// ---------------------------------------------------------------------------
// WMMA helpers: LDS fragment gathers matching ISA 7.12.2 layouts (wave32).
// A 16x32 bf16: lanes 0-15 hold M=lane, K = {0..7, 16..23}; lanes 16-31: K = {8..15, 24..31}.
// B 32x16 bf16: lanes 0-15 hold N=lane, K = {0..7, 16..23}; lanes 16-31: K = {8..15, 24..31}.
// LDS tiles stored row-major [128][32] bf16 (64B rows).
// ---------------------------------------------------------------------------
__device__ __forceinline__ v16bf load_frag_a(const __bf16* lds, int baseRow, int lane)
{
  int r = baseRow + (lane & 15);
  int half = lane >> 4;
  const __bf16* p = lds + r * 32 + half * 8;
  union { v16bf v; uint4 u[2]; } f;
  f.u[0] = *(const uint4*)(p);        // K = half*8 .. +7
  f.u[1] = *(const uint4*)(p + 16);   // K = 16+half*8 .. +7
  return f.v;
}
__device__ __forceinline__ v16bf load_frag_b(const __bf16* lds, int baseRow, int lane)
{
  int r = baseRow + (lane & 15);
  int half = lane >> 4;
  const __bf16* p = lds + r * 32 + half * 8;
  union { v16bf v; uint4 u[2]; } f;
  f.u[0] = *(const uint4*)(p);        // K = half*8 .. +7
  f.u[1] = *(const uint4*)(p + 16);   // K = 16+half*8 .. +7
  return f.v;
}
__device__ __forceinline__ v8f wmma_bf16(v16bf a, v16bf b, v8f c)
{
  return __builtin_amdgcn_wmma_f32_16x16x32_bf16(false, a, false, b, (short)0, c, false, false);
}

// ---------------------------------------------------------------------------
// Kernel 4: fused distance-GEMM + dual segment-max.
// Block = 256 threads (8 waves), tile 128x128, each wave owns 32x64 (2x4 frags).
// Double-buffered LDS staging via async global->LDS copies (ASYNCcnt), so the
// step-(k+1) copy overlaps the step-k WMMA stream.
// Epilogue: em = -sqrt(max(a2+b2-2acc,1e-12)) -> LDS, then run-length
// segment-max flushes via unsigned atomicMin (all em < 0 => order-free max).
// ---------------------------------------------------------------------------
struct Tiles { __bf16 a_hi[TILE*32], a_lo[TILE*32], b_hi[TILE*32], b_lo[TILE*32]; }; // 32 KB

__global__ __launch_bounds__(256) void edge_gemm_tile(
    const __bf16* __restrict__ efn_hi, const __bf16* __restrict__ efn_lo,
    const __bf16* __restrict__ efl_hi, const __bf16* __restrict__ efl_lo,
    const float* __restrict__ sqn, const float* __restrict__ sql,
    const int* __restrict__ node_seg, const int* __restrict__ label_seg,
    unsigned* __restrict__ rowmax, unsigned* __restrict__ colmax)
{
  __shared__ union {
    Tiles t[2];                 // 64 KB (double buffer)
    float em[TILE * EMP];       // 67.5 KB
  } sm;
  __shared__ int   s_nseg[TILE], s_lseg[TILE];
  __shared__ float s_an[TILE], s_bn[TILE];

  const int tid  = threadIdx.x;
  const int lane = tid & 31;
  const int wave = tid >> 5;
  const int r0 = blockIdx.y * TILE;
  const int c0 = blockIdx.x * TILE;

  if (tid < TILE) {
    s_nseg[tid] = node_seg[r0 + tid];
    s_lseg[tid] = label_seg[c0 + tid];
    s_an[tid]   = sqn[r0 + tid];
    s_bn[tid]   = sql[c0 + tid];
  }

  const int mBase = (wave & 3) * 32;   // 4 row groups of 32
  const int nBase = (wave >> 2) * 64;  // 2 col groups of 64

  // Stage one K=32 panel (4 matrices) into buffer bi: 8 async b128 per thread.
  auto stage = [&](int kk, int bi) {
    #pragma unroll
    for (int c = tid; c < TILE * 4; c += 256) {
      int row = c >> 2, ch = (c & 3) * 8;
      size_t ga = (size_t)(r0 + row) * DIMD + kk + ch;
      size_t gb = (size_t)(c0 + row) * DIMD + kk + ch;
      async_b128(efn_hi + ga, &sm.t[bi].a_hi[row * 32 + ch]);
      async_b128(efn_lo + ga, &sm.t[bi].a_lo[row * 32 + ch]);
      async_b128(efl_hi + gb, &sm.t[bi].b_hi[row * 32 + ch]);
      async_b128(efl_lo + gb, &sm.t[bi].b_lo[row * 32 + ch]);
    }
  };

  v8f acc[2][4];
  #pragma unroll
  for (int i = 0; i < 2; i++)
    #pragma unroll
    for (int j = 0; j < 4; j++)
      #pragma unroll
      for (int k = 0; k < 8; k++) acc[i][j][k] = 0.f;

  int p = 0;
  stage(0, 0);
  for (int kk = 0; kk < DIMD; kk += 32) {
    if (kk + 32 < DIMD) {
      stage(kk + 32, p ^ 1);   // overlaps with this step's WMMAs
      wait_async<8>();         // own wave's step-kk copies (in-order) complete
    } else {
      wait_async<0>();
    }
    __syncthreads();           // publish buffer p to all waves

    v16bf ah[2], al[2];
    #pragma unroll
    for (int fr = 0; fr < 2; fr++) {
      ah[fr] = load_frag_a(sm.t[p].a_hi, mBase + fr * 16, lane);
      al[fr] = load_frag_a(sm.t[p].a_lo, mBase + fr * 16, lane);
    }
    #pragma unroll
    for (int fc = 0; fc < 4; fc++) {
      v16bf bh = load_frag_b(sm.t[p].b_hi, nBase + fc * 16, lane);
      v16bf bl = load_frag_b(sm.t[p].b_lo, nBase + fc * 16, lane);
      #pragma unroll
      for (int fr = 0; fr < 2; fr++) {
        acc[fr][fc] = wmma_bf16(ah[fr], bh, acc[fr][fc]);
        acc[fr][fc] = wmma_bf16(ah[fr], bl, acc[fr][fc]);
        acc[fr][fc] = wmma_bf16(al[fr], bh, acc[fr][fc]);
      }
    }
    __syncthreads();           // everyone done reading buffer p
    p ^= 1;
  }

  // Epilogue: distances into LDS tile. D layout: lane holds N=lane&15,
  // rows m = (lane>>4)*8 + r for VGPR r.
  {
    const int nloc = lane & 15, half = lane >> 4;
    #pragma unroll
    for (int fr = 0; fr < 2; fr++) {
      #pragma unroll
      for (int fc = 0; fc < 4; fc++) {
        int mrow0 = mBase + fr * 16 + half * 8;
        int ncol  = nBase + fc * 16 + nloc;
        float bn = s_bn[ncol];
        #pragma unroll
        for (int r = 0; r < 8; r++) {
          int m = mrow0 + r;
          float d2 = s_an[m] + bn - 2.0f * acc[fr][fc][r];
          sm.em[m * EMP + ncol] = -__builtin_sqrtf(fmaxf(d2, 1e-12f));
        }
      }
    }
  }
  __syncthreads();

  // rowmax[e][label_seg]: 2 threads per row, 64 cols each; run-length flush.
  {
    int row = tid >> 1, cbeg = (tid & 1) * 64;
    float run = NEG_INF_F;
    int cur = s_lseg[cbeg];
    #pragma unroll 4
    for (int i = 0; i < 64; i++) {
      int cc = cbeg + i, sg = s_lseg[cc];
      if (sg != cur) {
        atomicMin(&rowmax[(size_t)(r0 + row) * BL + cur], f_as_u(run));
        run = NEG_INF_F; cur = sg;
      }
      run = fmaxf(run, sm.em[row * EMP + cc]);
    }
    atomicMin(&rowmax[(size_t)(r0 + row) * BL + cur], f_as_u(run));
  }
  // colmax[node_seg][j]: 2 threads per col, 64 rows each.
  {
    int col = tid >> 1, rbeg = (tid & 1) * 64;
    float run = NEG_INF_F;
    int cur = s_nseg[rbeg];
    #pragma unroll 4
    for (int i = 0; i < 64; i++) {
      int rr = rbeg + i, sg = s_nseg[rr];
      if (sg != cur) {
        atomicMin(&colmax[(size_t)cur * EL + (c0 + col)], f_as_u(run));
        run = NEG_INF_F; cur = sg;
      }
      run = fmaxf(run, sm.em[rr * EMP + col]);
    }
    atomicMin(&colmax[(size_t)cur * EL + (c0 + col)], f_as_u(run));
  }
}

// ---------------------------------------------------------------------------
// Kernel 5: out_n[b][l] = mean over node-segment rows of rowmax (deterministic).
// ---------------------------------------------------------------------------
__global__ __launch_bounds__(256) void out_n_kernel(
    const unsigned* __restrict__ rowmax,
    const int* __restrict__ startN, const int* __restrict__ cntN,
    float* __restrict__ out_n)
{
  int b = blockIdx.x, t = threadIdx.x;
  int s = startN[b], n = cntN[b];
  float inv = 1.0f / fmaxf((float)n, 1.0f);
  for (int l = t; l < BL; l += 256) {
    float sum = 0.f;
    for (int e = s; e < s + n; e++) sum += u_as_f(rowmax[(size_t)e * BL + l]);
    out_n[(size_t)b * BL + l] = sum * inv;
  }
}

// ---------------------------------------------------------------------------
// Kernel 6: out_l[b][l] = mean over label-segment cols of colmax (deterministic).
// ---------------------------------------------------------------------------
__global__ __launch_bounds__(256) void out_l_kernel(
    const unsigned* __restrict__ colmax,
    const int* __restrict__ startL, const int* __restrict__ cntL,
    float* __restrict__ out_l)
{
  int b = blockIdx.x, t = threadIdx.x;
  for (int l = t; l < BL; l += 256) {
    int s = startL[l], n = cntL[l];
    float sum = 0.f;
    for (int j = s; j < s + n; j++) sum += u_as_f(colmax[(size_t)b * EL + j]);
    out_l[(size_t)b * BL + l] = sum / fmaxf((float)n, 1.0f);
  }
}

// ---------------------------------------------------------------------------
// Kernel 7: node-level cdist + blend + top-10 (lower-index tie-break like jax).
// ---------------------------------------------------------------------------
__global__ __launch_bounds__(256) void combine_topk(
    const float* __restrict__ emb,
    const int* __restrict__ node_ids, const int* __restrict__ label_ids,
    const float* __restrict__ out_n, const float* __restrict__ out_l,
    float* __restrict__ score_out, int* __restrict__ idx_out)
{
  __shared__ __align__(16) float qrow[DIMD];
  __shared__ float s[BL];
  __shared__ float rv[256];
  __shared__ int   ri[256];
  __shared__ float qn2s;

  int b = blockIdx.x, t = threadIdx.x;
  qrow[t] = emb[(size_t)node_ids[b] * DIMD + t];
  __syncthreads();
  rv[t] = qrow[t] * qrow[t];
  __syncthreads();
  for (int off = 128; off > 0; off >>= 1) { if (t < off) rv[t] += rv[t + off]; __syncthreads(); }
  if (t == 0) qn2s = rv[0];
  __syncthreads();
  float qn2 = qn2s;

  for (int l = t; l < BL; l += 256) {
    const float* lr = emb + (size_t)label_ids[l] * DIMD;
    float dot = 0.f, b2 = 0.f;
    #pragma unroll 4
    for (int d = 0; d < DIMD; d += 4) {
      float4 lv = *(const float4*)(lr + d);
      float4 qv = *(const float4*)(qrow + d);
      dot += qv.x * lv.x + qv.y * lv.y + qv.z * lv.z + qv.w * lv.w;
      b2  += lv.x * lv.x + lv.y * lv.y + lv.z * lv.z + lv.w * lv.w;
    }
    float ns = -__builtin_sqrtf(fmaxf(qn2 + b2 - 2.f * dot, 1e-12f));
    float neigh = 0.5f * (out_n[(size_t)b * BL + l] + out_l[(size_t)b * BL + l]);
    float sc = (ALPHA_C * ns + (1.0f - ALPHA_C) * neigh) / TEMP_C;
    s[l] = sc;
    score_out[(size_t)b * BL + l] = sc;
  }
  __syncthreads();

  for (int k = 0; k < KTOP; k++) {
    float v1 = s[t]; int i1 = t;
    float v2 = s[t + 256];
    if (v2 > v1) { v1 = v2; i1 = t + 256; }   // tie -> keep lower index
    rv[t] = v1; ri[t] = i1;
    __syncthreads();
    for (int off = 128; off > 0; off >>= 1) {
      if (t < off) {
        float vo = rv[t + off]; int io = ri[t + off];
        if (vo > rv[t] || (vo == rv[t] && io < ri[t])) { rv[t] = vo; ri[t] = io; }
      }
      __syncthreads();
    }
    if (t == 0) { idx_out[b * KTOP + k] = ri[0]; s[ri[0]] = NEG_INF_F; }
    __syncthreads();
  }
}

// ---------------------------------------------------------------------------
extern "C" void kernel_launch(void* const* d_in, const int* in_sizes, int n_in,
                              void* d_out, int out_size, void* d_ws, size_t ws_size,
                              hipStream_t stream)
{
  const float* emb       = (const float*)d_in[0];
  const int*   node_ids  = (const int*)d_in[1];
  const int*   label_ids = (const int*)d_in[2];
  const int*   node_edge = (const int*)d_in[3];
  const int*   label_edge= (const int*)d_in[4];
  const int*   node_seg  = (const int*)d_in[5];
  const int*   label_seg = (const int*)d_in[6];

  char* ws = (char*)d_ws;
  size_t off = 0;
  auto take = [&](size_t bytes) -> char* {
    char* p = ws + off;
    off += (bytes + 255) & ~(size_t)255;
    return p;
  };
  __bf16* efn_hi = (__bf16*)take((size_t)EN * DIMD * 2);
  __bf16* efn_lo = (__bf16*)take((size_t)EN * DIMD * 2);
  __bf16* efl_hi = (__bf16*)take((size_t)EL * DIMD * 2);
  __bf16* efl_lo = (__bf16*)take((size_t)EL * DIMD * 2);
  float*  sqn    = (float*)take((size_t)EN * 4);
  float*  sql    = (float*)take((size_t)EL * 4);
  int* startN = (int*)take(BN * 4);
  int* cntN   = (int*)take(BN * 4);
  int* startL = (int*)take(BL * 4);
  int* cntL   = (int*)take(BL * 4);
  unsigned* rowmax = (unsigned*)take((size_t)EN * BL * 4);   // multiple of 256 ->
  unsigned* colmax = (unsigned*)take((size_t)BN * EL * 4);   // contiguous with rowmax
  float* out_n_buf = (float*)take((size_t)BN * BL * 4);
  float* out_l_buf = (float*)take((size_t)BN * BL * 4);

  // 1. edge features (bf16 hi/lo split) + norms
  build_edge_feats<<<(EN + EL) / 8, 256, 0, stream>>>(
      emb, node_edge, label_edge, efn_hi, efn_lo, efl_hi, efl_lo, sqn, sql);

  // 2. segment bounds
  seg_bounds<<<2, 512, 0, stream>>>(node_seg, label_seg, startN, cntN, startL, cntL);

  // 3. -inf init of both max buffers (contiguous)
  int n4 = (int)(((size_t)EN * BL + (size_t)BN * EL) / 4);
  init_neginf<<<3072, 256, 0, stream>>>((uint4*)rowmax, n4);

  // 4. fused WMMA distance GEMM + segment maxes (async double-buffered staging)
  edge_gemm_tile<<<dim3(EL / TILE, EN / TILE), 256, 0, stream>>>(
      efn_hi, efn_lo, efl_hi, efl_lo, sqn, sql, node_seg, label_seg, rowmax, colmax);

  // 5/6. segment means
  out_n_kernel<<<BN, 256, 0, stream>>>(rowmax, startN, cntN, out_n_buf);
  out_l_kernel<<<BN, 256, 0, stream>>>(colmax, startL, cntL, out_l_buf);

  // 7. node cdist + blend + top-k
  float* score_out = (float*)d_out;
  int*   idx_out   = ((int*)d_out) + (size_t)BN * BL;
  combine_topk<<<BN, 256, 0, stream>>>(
      emb, node_ids, label_ids, out_n_buf, out_l_buf, score_out, idx_out);
}
